// SelfSupervisedMamba2_33921651704435
// MI455X (gfx1250) — matrix operand order repaired and checked by reading, not compiled, not run on hardware
//
#include <hip/hip_runtime.h>
#include <hip/hip_bf16.h>

// ---------------- Mamba2 block constants (from reference) ----------------
#define DM      256      // d_model
#define DIP     1160     // d_in_proj
#define DI      512      // d_inner
#define DS      64       // d_state
#define NH      8        // heads
#define HD      64       // headdim
#define CONVD   640      // conv dim
#define DCONV   4
#define CH      256      // chunk
#define NCHUNK  16       // L / CH
#define BATCH   8
#define SEQ     4096
#define BL      (BATCH * SEQ)   // 32768 token rows
#define EPS     1e-5f

typedef __attribute__((ext_vector_type(16))) __bf16 v16bf;
typedef __attribute__((ext_vector_type(8)))  float  v8f;

// ---------------- WMMA fragment helpers (CDNA5 16x16x32 bf16 layouts) ----
// A-matrix 16x32 (MxK), 16-bit: lane L (row = L&15, group g = L>>4),
// element e -> K = 16g + e (contiguous k order chosen here; B frags use the same).
__device__ __forceinline__ int amap(int e, int g) {
  return 16 * g + e;
}

__device__ __forceinline__ v8f wmma_bf16(v16bf a, v16bf b, v8f c) {
  return __builtin_amdgcn_wmma_f32_16x16x32_bf16(false, a, false, b, (short)0, c,
                                                 false, false);
}

// A-frag from row-major fp32 MxK matrix
__device__ __forceinline__ v16bf load_a_frag(const float* __restrict__ A, int lda,
                                             int row0, int k0) {
  int lane = threadIdx.x & 31;
  int g = lane >> 4;
  const float* row = A + (size_t)(row0 + (lane & 15)) * lda + k0;
  v16bf a;
#pragma unroll
  for (int e = 0; e < 16; ++e) a[e] = (__bf16)row[amap(e, g)];
  return a;
}

// B-frag (KxN row-major): lane col = col0+(L&15); K = 16g + e (strided loads)
__device__ __forceinline__ v16bf load_b_frag_stride(const float* __restrict__ P,
                                                    int ld, int k0, int col0) {
  int lane = threadIdx.x & 31;
  int g = lane >> 4;
  const float* p = P + (size_t)(k0 + 16 * g) * ld + col0 + (lane & 15);
  v16bf b;
#pragma unroll
  for (int e = 0; e < 16; ++e) b[e] = (__bf16)p[(size_t)e * ld];
  return b;
}

// B-frag where logical (k, col) lives at P[col*ld + k]  (i.e. B^T from a
// row-major matrix) -> 16 contiguous floats per lane (vectorizes to b128 loads)
__device__ __forceinline__ v16bf load_bt_frag(const float* __restrict__ P, int ld,
                                              int k0, int col0) {
  int lane = threadIdx.x & 31;
  int g = lane >> 4;
  const float* row = P + (size_t)(col0 + (lane & 15)) * ld + k0 + 16 * g;
  v16bf b;
#pragma unroll
  for (int e = 0; e < 16; ++e) b[e] = (__bf16)row[e];
  return b;
}

// ---------------- CDNA5 async-DMA (ASYNCcnt) helpers ----------------
__device__ __forceinline__ unsigned lds_addr_of(const void* p) {
  // generic pointer to a __shared__ object: low 32 bits are the LDS byte addr
  return (unsigned)(unsigned long long)p;
}
__device__ __forceinline__ void async_g2lds_b128(unsigned lds_off,
                                                 const float* gaddr) {
  asm volatile("global_load_async_to_lds_b128 %0, %1, off"
               :: "v"(lds_off), "v"(gaddr)
               : "memory");
}
__device__ __forceinline__ void wait_asynccnt_le2() {  // prev panel retired
  asm volatile("s_wait_asynccnt 0x2" ::: "memory");
}
__device__ __forceinline__ void wait_asynccnt_le0() {  // drain
  asm volatile("s_wait_asynccnt 0x0" ::: "memory");
}

// Issue one 32x64 fp32 B panel as async DMA: 512 x 16B chunks, 2 per thread,
// i.e. exactly 2 ASYNCcnt ops per wave (uniform -> counted waits are exact).
// Ragged columns load from a clamped in-bounds 16B-aligned address and are
// zeroed after the wait (see fixup in the kernel).
__device__ __forceinline__ void stage_panel(const float* __restrict__ B, int N,
                                            int n0, int k0, unsigned ldsbuf,
                                            int tid) {
#pragma unroll
  for (int c2 = 0; c2 < 2; ++c2) {
    int chunk = tid * 2 + c2;            // 0..511
    int kk = chunk >> 4;                 // panel row (16 x 16B = 64 floats/row)
    int gcol = n0 + (chunk & 15) * 4;    // first float col of this chunk
    int gc = (gcol + 3 < N) ? gcol : (N - 4);
    async_g2lds_b128(ldsbuf + (unsigned)(chunk * 16),
                     B + (size_t)(k0 + kk) * N + gc);
  }
}

// ------- Blocked WMMA GEMM: C(MxN) = A(MxK) @ B(KxN) [+bias] -------------
// Block = 256 threads = 8 waves -> 128 rows x 64 cols of C.
// Double-buffered ASYNC DMA pipeline: while the 8 waves run 4 WMMAs each on
// panel k (8x LDS reuse, 4x A-frag reuse), panel k+1 streams into the other
// 8 KB LDS buffer; s_wait_asynccnt 2 retires exactly the previous panel.
template <int Kdim>
__global__ __launch_bounds__(256) void wmma_gemm2_kernel(
    const float* __restrict__ A, const float* __restrict__ B,
    float* __restrict__ C, const float* __restrict__ bias, int M, int N) {
  constexpr int BN = 64;
  constexpr int BK = 32;
  constexpr int NK = Kdim / BK;
  __shared__ float sB[2][BK * BN];       // 2 x 8 KB fp32 panels

  int nt = (N + BN - 1) / BN;
  int bm = blockIdx.x / nt, bn = blockIdx.x % nt;
  int m0 = bm * 128, n0 = bn * BN;

  int tid = threadIdx.x, w = tid >> 5, lane = tid & 31;
  int lrow = lane & 15, g = lane >> 4;
  int mrow = m0 + 16 * w;

  unsigned sb0 = lds_addr_of(&sB[0][0]);
  bool ragged = (n0 + BN > N);

  v8f acc[4] = {v8f{}, v8f{}, v8f{}, v8f{}};

  stage_panel(B, N, n0, 0, sb0, tid);    // prologue: panel 0 -> buffer 0

#pragma unroll
  for (int ks = 0; ks < NK; ++ks) {
    int cur = ks & 1;
    if (ks + 1 < NK) {
      stage_panel(B, N, n0, (ks + 1) * BK,
                  sb0 + (unsigned)((cur ^ 1) * BK * BN * 4), tid);
      wait_asynccnt_le2();               // exactly panel ks has landed
    } else {
      wait_asynccnt_le0();
    }
    if (ragged) {                        // zero the out-of-range columns
#pragma unroll
      for (int c2 = 0; c2 < 2; ++c2) {
        int chunk = tid * 2 + c2;
        int gcol = n0 + (chunk & 15) * 4;
#pragma unroll
        for (int q = 0; q < 4; ++q)
          if (gcol + q >= N) sB[cur][chunk * 4 + q] = 0.f;
      }
    }
    __syncthreads();

    // one A-frag (global, L2-resident) feeds 4 WMMAs against the LDS panel
    v16bf a = load_a_frag(A, Kdim, mrow, ks * BK);
#pragma unroll
    for (int pt = 0; pt < 4; ++pt) {
      v16bf bfr;
#pragma unroll
      for (int e = 0; e < 16; ++e)
        bfr[e] = (__bf16)sB[cur][(16 * g + e) * BN + 16 * pt + lrow];
      acc[pt] = wmma_bf16(a, bfr, acc[pt]);
    }
    __syncthreads();
  }

#pragma unroll
  for (int pt = 0; pt < 4; ++pt) {
    int n = n0 + 16 * pt + lrow;
    if (n >= N) continue;
    float bv = bias ? bias[n] : 0.f;
#pragma unroll
    for (int i = 0; i < 8; ++i)
      C[(size_t)(mrow + 8 * g + i) * N + n] = acc[pt][i] + bv;
  }
}

// ---------------- depthwise causal conv1d + SiLU + split ----------------
__global__ __launch_bounds__(256) void conv_kernel(
    const float* __restrict__ zx, const float* __restrict__ cw,
    const float* __restrict__ cb, float* __restrict__ xhb,
    float* __restrict__ Bcb, float* __restrict__ Ccb) {
  int idx = blockIdx.x * 256 + threadIdx.x;
  if (idx >= BL * CONVD) return;
  int ch = idx % CONVD, bl = idx / CONVD;
  int b = bl / SEQ, l = bl % SEQ;
  float acc = cb[ch];
#pragma unroll
  for (int k = 0; k < DCONV; ++k) {
    int ls = l - (DCONV - 1) + k;
    if (ls >= 0) acc += zx[(size_t)(b * SEQ + ls) * DIP + DI + ch] * cw[ch * DCONV + k];
  }
  float v = acc / (1.f + __expf(-acc));  // SiLU
  if (ch < DI)            xhb[(size_t)bl * DI + ch] = v;
  else if (ch < DI + DS)  Bcb[(size_t)bl * DS + (ch - DI)] = v;
  else                    Ccb[(size_t)bl * DS + (ch - DI - DS)] = v;
}

// ---- softplus(dt)+bias, dA = dt*A, per-chunk inclusive cumsum (segsum base) --
__global__ __launch_bounds__(CH) void dt_scan_kernel(
    const float* __restrict__ zx, const float* __restrict__ dt_bias,
    const float* __restrict__ A_log, float* __restrict__ dtc,
    float* __restrict__ Acum) {
  int blk = blockIdx.x;                // ((b*NCHUNK)+c)*NH + h
  int h = blk % NH;
  int bc = blk / NH;
  int c = bc % NCHUNK, b = bc / NCHUNK;
  int l = threadIdx.x;
  int row = b * SEQ + c * CH + l;
  float xv = zx[(size_t)row * DIP + (DI + CONVD) + h] + dt_bias[h];
  float dt = (xv > 20.f) ? xv : log1pf(__expf(xv));  // softplus
  dtc[(size_t)row * NH + h] = dt;
  float da = dt * (-__expf(A_log[h]));
  __shared__ float s[CH];
  s[l] = da;
  __syncthreads();
  for (int off = 1; off < CH; off <<= 1) {   // Hillis-Steele inclusive scan
    float v = (l >= off) ? s[l - off] : 0.f;
    __syncthreads();
    s[l] += v;
    __syncthreads();
  }
  Acum[(size_t)blk * CH + l] = s[l];
}

// ---------------- xdt = xh * dt ----------------
__global__ __launch_bounds__(256) void xdt_kernel(const float* __restrict__ xh,
                                                  const float* __restrict__ dtc,
                                                  float* __restrict__ xdt) {
  size_t idx = (size_t)blockIdx.x * 256 + threadIdx.x;
  if (idx >= (size_t)BL * DI) return;
  size_t row = idx >> 9;
  int j = (int)(idx & 511);
  xdt[idx] = xh[idx] * dtc[row * NH + (j >> 6)];
}

// ------- intra-chunk Y_diag (+ xh*D) and per-chunk state, one (b,c,h)/block ---
// 16 waves: wave w owns rows [16w,16w+16) of the 256x256 masked score matrix
// and one 16x16 tile of the 64x64 state. All matmuls via WMMA bf16.
__global__ __launch_bounds__(512) void chunk_diag_states_kernel(
    const float* __restrict__ BcB, const float* __restrict__ CcB,
    const float* __restrict__ xdt, const float* __restrict__ xh,
    const float* __restrict__ Acum, const float* __restrict__ Dp,
    float* __restrict__ Y, float* __restrict__ states) {
  int blk = blockIdx.x;
  int h = blk % NH;
  int bc = blk / NH;
  int c = bc % NCHUNK, b = bc / NCHUNK;
  size_t base = (size_t)b * SEQ + (size_t)c * CH;
  int tid = threadIdx.x, w = tid >> 5, lane = tid & 31;
  int lrow = lane & 15, g = lane >> 4;

  __shared__ float  sAcum[CH];
  __shared__ __bf16 sS[16][16][32];  // per-wave masked-score staging (bf16)

  if (tid < CH) sAcum[tid] = Acum[(size_t)blk * CH + tid];
  __syncthreads();

  const float* Cp  = CcB + base * DS;
  const float* Bp  = BcB + base * DS;
  const float* Xp  = xdt + base * DI + h * HD;
  const float* XHp = xh  + base * DI + h * HD;

  int l0 = 16 * w;
  // GEMM1 A-frags: C strip (16 x 64), fixed for the whole strip
  v16bf ca0 = load_a_frag(Cp, DS, l0, 0);
  v16bf ca1 = load_a_frag(Cp, DS, l0, 32);

  v8f accY[4] = {v8f{}, v8f{}, v8f{}, v8f{}};

  int tmax = w >> 1;                       // causal: only s-tiles with s0 <= l0+15
  for (int t = 0; t <= tmax; ++t) {
    int sbase = 32 * t;
#pragma unroll
    for (int half = 0; half < 2; ++half) {
      int s0 = sbase + 16 * half;
      // S_tile = C_strip @ B^T  (K = d_state = 64)
      v16bf bt0 = load_bt_frag(Bp, DS, 0, s0);
      v16bf bt1 = load_bt_frag(Bp, DS, 32, s0);
      v8f sacc = {};
      sacc = wmma_bf16(ca0, bt0, sacc);
      sacc = wmma_bf16(ca1, bt1, sacc);
      // mask + segment-sum decay in D-frag registers, stage to LDS as bf16
      int scol = s0 + lrow;
      float as = sAcum[scol];
#pragma unroll
      for (int i = 0; i < 8; ++i) {
        int l = l0 + 8 * g + i;
        float v = (scol <= l) ? sacc[i] * __expf(sAcum[l] - as) : 0.f;
        sS[w][8 * g + i][16 * half + lrow] = (__bf16)v;
      }
    }
    // wave-private LDS round trip: DS ops of one wave are in-order; just drain
    asm volatile("s_wait_dscnt 0x0" ::: "memory");
    // reload as A-frag (16 x 32 over s) and multiply into xdt (s x 64)
    v16bf afr;
#pragma unroll
    for (int e = 0; e < 16; ++e) afr[e] = sS[w][lrow][amap(e, g)];
#pragma unroll
    for (int pt = 0; pt < 4; ++pt) {
      v16bf bx = load_b_frag_stride(Xp, DI, sbase, 16 * pt);
      accY[pt] = wmma_bf16(afr, bx, accY[pt]);
    }
  }
  // Y = Y_diag + D*xh   (Y_off added later)
  float Dh = Dp[h];
#pragma unroll
  for (int pt = 0; pt < 4; ++pt) {
#pragma unroll
    for (int i = 0; i < 8; ++i) {
      int l = l0 + 8 * g + i;
      int p = 16 * pt + lrow;
      Y[(base + l) * DI + h * HD + p] = accY[pt][i] + Dh * XHp[(size_t)l * DI + p];
    }
  }

  // ---- chunk state: states[n,p] = sum_s B[s,n]*exp(Acum[last]-Acum[s])*xdt[s,p]
  int n0 = (w >> 2) * 16, p0 = (w & 3) * 16;
  float alast = sAcum[CH - 1];
  v8f sst = {};
  for (int k0 = 0; k0 < CH; k0 += 32) {
    v16bf a;
#pragma unroll
    for (int e = 0; e < 16; ++e) {
      int s = k0 + amap(e, g);
      a[e] = (__bf16)(Bp[(size_t)s * DS + n0 + lrow] * __expf(alast - sAcum[s]));
    }
    v16bf bx = load_b_frag_stride(Xp, DI, k0, p0);
    sst = wmma_bf16(a, bx, sst);
  }
  float* st = states + (size_t)blk * (DS * HD);
#pragma unroll
  for (int i = 0; i < 8; ++i)
    st[(size_t)(n0 + 8 * g + i) * HD + p0 + lrow] = sst[i];
}

// ---------------- sequential inter-chunk scan (tiny) ----------------
__global__ __launch_bounds__(256) void state_scan_kernel(
    const float* __restrict__ states, const float* __restrict__ Acum,
    float* __restrict__ prevs) {
  int b = blockIdx.x / NH, h = blockIdx.x % NH;
  int t = threadIdx.x;
  float hreg[16];
#pragma unroll
  for (int i = 0; i < 16; ++i) hreg[i] = 0.f;
  for (int c = 0; c < NCHUNK; ++c) {
    size_t blk = (size_t)(b * NCHUNK + c) * NH + h;
    size_t off = blk * (DS * HD) + (size_t)t * 16;
    float dec = __expf(Acum[blk * CH + CH - 1]);  // chunk_decay
#pragma unroll
    for (int i = 0; i < 16; ++i) {
      prevs[off + i] = hreg[i];                   // state entering chunk c
      hreg[i] = hreg[i] * dec + states[off + i];
    }
  }
}

// ------- Y += (C * exp(Acum)) @ prevs^T   per (b,c,h), WMMA K=64 -------
__global__ __launch_bounds__(512) void chunk_off_kernel(
    const float* __restrict__ CcB, const float* __restrict__ prevs,
    const float* __restrict__ Acum, float* __restrict__ Y) {
  int blk = blockIdx.x;
  int h = blk % NH;
  int bc = blk / NH;
  int c = bc % NCHUNK, b = bc / NCHUNK;
  size_t base = (size_t)b * SEQ + (size_t)c * CH;
  int tid = threadIdx.x, w = tid >> 5, lane = tid & 31;
  int lrow = lane & 15, g = lane >> 4;

  __shared__ float sAcum[CH];
  if (tid < CH) sAcum[tid] = Acum[(size_t)blk * CH + tid];
  __syncthreads();

  const float* Cp = CcB + base * DS;
  const float* Pv = prevs + (size_t)blk * (DS * HD);  // [n][p]
  int l0 = 16 * w;

  float el = __expf(sAcum[l0 + lrow]);          // row decay, same for all e
  const float* crow = Cp + (size_t)(l0 + lrow) * DS;
  v16bf a0, a1;
#pragma unroll
  for (int e = 0; e < 16; ++e) {
    a0[e] = (__bf16)(crow[amap(e, g)] * el);
    a1[e] = (__bf16)(crow[32 + amap(e, g)] * el);
  }
#pragma unroll
  for (int pt = 0; pt < 4; ++pt) {
    v16bf b0 = load_b_frag_stride(Pv, HD, 0, 16 * pt);
    v16bf b1 = load_b_frag_stride(Pv, HD, 32, 16 * pt);
    v8f acc = {};
    acc = wmma_bf16(a0, b0, acc);
    acc = wmma_bf16(a1, b1, acc);
#pragma unroll
    for (int i = 0; i < 8; ++i) {
      int l = l0 + 8 * g + i, p = 16 * pt + lrow;
      Y[(base + l) * DI + h * HD + p] += acc[i];
    }
  }
}

// ---------------- gated RMSNorm: Y *= silu(z); rmsnorm(512) * norm_w --------
__global__ __launch_bounds__(256) void gate_rms_kernel(
    const float* __restrict__ Y, const float* __restrict__ zx,
    const float* __restrict__ norm_w, float* __restrict__ Yn) {
  int row = blockIdx.x, t = threadIdx.x;
  float y[2];
  float ss = 0.f;
#pragma unroll
  for (int k = 0; k < 2; ++k) {
    int j = t + 256 * k;
    float z = zx[(size_t)row * DIP + j];
    float v = Y[(size_t)row * DI + j] * (z / (1.f + __expf(-z)));
    y[k] = v;
    ss += v * v;
  }
  __shared__ float red[256];
  red[t] = ss;
  __syncthreads();
  for (int s2 = 128; s2 > 0; s2 >>= 1) {
    if (t < s2) red[t] += red[t + s2];
    __syncthreads();
  }
  float scale = rsqrtf(red[0] / (float)DI + EPS);
#pragma unroll
  for (int k = 0; k < 2; ++k) {
    int j = t + 256 * k;
    Yn[(size_t)row * DI + j] = y[k] * scale * norm_w[j];
  }
}

// ---------------- host-side orchestration ----------------
extern "C" void kernel_launch(void* const* d_in, const int* in_sizes, int n_in,
                              void* d_out, int out_size, void* d_ws,
                              size_t ws_size, hipStream_t stream) {
  const float* x       = (const float*)d_in[0];
  const float* W_in    = (const float*)d_in[1];
  const float* conv_w  = (const float*)d_in[2];
  const float* conv_b  = (const float*)d_in[3];
  const float* dt_bias = (const float*)d_in[4];
  const float* A_log   = (const float*)d_in[5];
  const float* Dp      = (const float*)d_in[6];
  const float* norm_w  = (const float*)d_in[7];
  const float* W_out   = (const float*)d_in[8];
  const float* W_lin   = (const float*)d_in[9];
  const float* b_lin   = (const float*)d_in[10];
  float* out = (float*)d_out;

  float* ws = (float*)d_ws;
  size_t o = 0;
  float* zx    = ws + o; o += (size_t)BL * DIP;                      // 152 MB
  float* xhb   = ws + o; o += (size_t)BL * DI;
  float* xdtb  = ws + o; o += (size_t)BL * DI;
  float* Bcb   = ws + o; o += (size_t)BL * DS;
  float* Ccb   = ws + o; o += (size_t)BL * DS;
  float* dtcb  = ws + o; o += (size_t)BL * NH;
  float* acumb = ws + o; o += (size_t)BATCH * NCHUNK * NH * CH;
  float* Yb    = ws + o; o += (size_t)BL * DI;
  float* stb   = ws + o; o += (size_t)BATCH * NCHUNK * NH * DS * HD;
  float* pvb   = ws + o; o += (size_t)BATCH * NCHUNK * NH * DS * HD;
  float* Ynb   = ws + o; o += (size_t)BL * DI;
  float* o1b   = ws + o; o += (size_t)BL * DM;

  // 1) in_proj: zxbcdt = x @ W_in   (32768 x 256 @ 256 x 1160)
  {
    int nt = (DIP + 63) / 64;                 // 19 col-panels
    wmma_gemm2_kernel<DM><<<(BL / 128) * nt, 256, 0, stream>>>(x, W_in, zx,
                                                               nullptr, BL, DIP);
  }
  // 2) depthwise causal conv + SiLU, split into xh / B / C
  conv_kernel<<<(BL * CONVD + 255) / 256, 256, 0, stream>>>(zx, conv_w, conv_b,
                                                            xhb, Bcb, Ccb);
  // 3) softplus(dt)+bias, dA cumsum per chunk
  dt_scan_kernel<<<BATCH * NCHUNK * NH, CH, 0, stream>>>(zx, dt_bias, A_log,
                                                         dtcb, acumb);
  // 4) xdt = xh * dt
  xdt_kernel<<<(int)(((size_t)BL * DI + 255) / 256), 256, 0, stream>>>(xhb, dtcb,
                                                                       xdtb);
  // 5) intra-chunk diag contribution + per-chunk states (WMMA heavy)
  chunk_diag_states_kernel<<<BATCH * NCHUNK * NH, 512, 0, stream>>>(
      Bcb, Ccb, xdtb, xhb, acumb, Dp, Yb, stb);
  // 6) sequential inter-chunk scan
  state_scan_kernel<<<BATCH * NH, 256, 0, stream>>>(stb, acumb, pvb);
  // 7) inter-chunk contribution Y += (C*expA) @ prevs^T
  chunk_off_kernel<<<BATCH * NCHUNK * NH, 512, 0, stream>>>(Ccb, pvb, acumb, Yb);
  // 8) gated RMSNorm
  gate_rms_kernel<<<BL, 256, 0, stream>>>(Yb, zx, norm_w, Ynb);
  // 9) out_proj: o1 = Yn @ W_out  (K = 512)
  wmma_gemm2_kernel<DI><<<(BL / 128) * (DM / 64), 256, 0, stream>>>(
      Ynb, W_out, o1b, nullptr, BL, DM);
  // 10) final linear: out = o1 @ W_lin + b_lin  (K = 256)
  wmma_gemm2_kernel<DM><<<(BL / 128) * (DM / 64), 256, 0, stream>>>(
      o1b, W_lin, out, b_lin, BL, DM);
}
